// PointNet2SemSeg_PointConv_64063732187333
// MI455X (gfx1250) — matrix operand order repaired and checked by reading, not compiled, not run on hardware
//
#include <hip/hip_runtime.h>

typedef __attribute__((ext_vector_type(16))) _Float16 v16h;
typedef __attribute__((ext_vector_type(8)))  _Float16 v8h;
typedef __attribute__((ext_vector_type(8)))  float    v8f;

#define EPS_BN 1e-5f

#define SHUF16(lo, hi) __builtin_shufflevector(lo, hi, 0,1,2,3,4,5,6,7,8,9,10,11,12,13,14,15)

// ======================= point-ops kernels (VALU/LDS) =======================

__global__ void transpose_b3n_f32_kernel(const float* __restrict__ in,
                                         float* __restrict__ out, int Bn, int Nn)
{
  int i = blockIdx.x * blockDim.x + threadIdx.x;
  int total = Bn * Nn * 3;
  if (i >= total) return;
  int c = i % 3;
  int n = (i / 3) % Nn;
  int b = i / (3 * Nn);
  out[i] = in[((size_t)b * 3 + c) * Nn + n];
}

__global__ void transpose_b3n_f16_kernel(const float* __restrict__ in,
                                         _Float16* __restrict__ out, int Bn, int Nn)
{
  int i = blockIdx.x * blockDim.x + threadIdx.x;
  int total = Bn * Nn * 3;
  if (i >= total) return;
  int c = i % 3;
  int n = (i / 3) % Nn;
  int b = i / (3 * Nn);
  out[i] = (_Float16)in[((size_t)b * 3 + c) * Nn + n];
}

// one block per batch; sequential farthest-point sampling with LDS distance array
__global__ void fps_kernel(const float* __restrict__ xyz, int* __restrict__ fidx,
                           int Nn, int S)
{
  __shared__ float dist[8192];
  __shared__ float redv[256];
  __shared__ int   redi[256];
  __shared__ int   s_far;
  int b = blockIdx.x;
  int t = threadIdx.x;
  const float* X = xyz + (size_t)b * Nn * 3;
  for (int i = t; i < Nn; i += 256) dist[i] = 1e10f;
  if (t == 0) s_far = 0;
  __syncthreads();
  for (int s = 0; s < S; ++s) {
    int far = s_far;
    if (t == 0) fidx[b * S + s] = far;
    float cx = X[far * 3 + 0], cy = X[far * 3 + 1], cz = X[far * 3 + 2];
    float bm = -1.0f; int bi = 0x7fffffff;
    for (int i = t; i < Nn; i += 256) {
      float dx = X[i * 3 + 0] - cx;
      float dy = X[i * 3 + 1] - cy;
      float dz = X[i * 3 + 2] - cz;
      float d  = dx * dx + dy * dy + dz * dz;
      float nd = fminf(dist[i], d);
      dist[i] = nd;
      if (nd > bm) { bm = nd; bi = i; }
    }
    redv[t] = bm; redi[t] = bi;
    __syncthreads();
    for (int off = 128; off > 0; off >>= 1) {
      if (t < off) {
        float ov = redv[t + off]; int oi = redi[t + off];
        if (ov > redv[t] || (ov == redv[t] && oi < redi[t])) { redv[t] = ov; redi[t] = oi; }
      }
      __syncthreads();
    }
    if (t == 0) s_far = redi[0];
    __syncthreads();
  }
}

__global__ void gather3_kernel(const float* __restrict__ in, const int* __restrict__ idx,
                               float* __restrict__ out, int Nin, int S, int Bn)
{
  int i = blockIdx.x * blockDim.x + threadIdx.x;
  int total = Bn * S * 3;
  if (i >= total) return;
  int c = i % 3;
  int s = (i / 3) % S;
  int b = i / (3 * S);
  out[i] = in[((size_t)b * Nin + idx[b * S + s]) * 3 + c];
}

// one thread per query point: first `ns` indices (ascending) within radius, pad with first
__global__ void ball_query_kernel(const float* __restrict__ xyz,
                                  const float* __restrict__ new_xyz,
                                  int* __restrict__ gidx,
                                  int Nn, int S, float r2, int ns, int Bn)
{
  int q = blockIdx.x * blockDim.x + threadIdx.x;
  if (q >= Bn * S) return;
  int b = q / S;
  const float* X = xyz + (size_t)b * Nn * 3;
  float cx = new_xyz[q * 3 + 0], cy = new_xyz[q * 3 + 1], cz = new_xyz[q * 3 + 2];
  int* out = gidx + (size_t)q * ns;
  int cnt = 0, first = 0;
  for (int i = 0; i < Nn && cnt < ns; ++i) {
    float dx = X[i * 3 + 0] - cx;
    float dy = X[i * 3 + 1] - cy;
    float dz = X[i * 3 + 2] - cz;
    if (dx * dx + dy * dy + dz * dz <= r2) {
      if (cnt == 0) first = i;
      out[cnt++] = i;
    }
  }
  for (int j = cnt; j < ns; ++j) out[j] = first;
}

// build grouped feature rows (f16, stride K32, zero-padded): [xyz[g]-new_xyz | points[g]]
__global__ void group_kernel(const float* __restrict__ xyz, const _Float16* __restrict__ pts,
                             const float* __restrict__ new_xyz, const int* __restrict__ gidx,
                             _Float16* __restrict__ A, int Nin, int S, int ns, int C,
                             int K32, int Bn)
{
  int r = blockIdx.x * blockDim.x + threadIdx.x;
  int total = Bn * S * ns;
  if (r >= total) return;
  int j  = r % ns;
  int bs = r / ns;
  int b  = bs / S;
  int g  = gidx[(size_t)bs * ns + j];
  _Float16* row = A + (size_t)r * K32;
  row[0] = (_Float16)(xyz[((size_t)b * Nin + g) * 3 + 0] - new_xyz[bs * 3 + 0]);
  row[1] = (_Float16)(xyz[((size_t)b * Nin + g) * 3 + 1] - new_xyz[bs * 3 + 1]);
  row[2] = (_Float16)(xyz[((size_t)b * Nin + g) * 3 + 2] - new_xyz[bs * 3 + 2]);
  const _Float16* P = pts + ((size_t)b * Nin + g) * C;
  for (int c = 0; c < C; ++c) row[3 + c] = P[c];
  for (int c = 3 + C; c < K32; ++c) row[c] = (_Float16)0.0f;
}

__global__ void maxpool_kernel(const _Float16* __restrict__ h, _Float16* __restrict__ out,
                               int R, int ns, int C)
{
  int i = blockIdx.x * blockDim.x + threadIdx.x;
  if (i >= R * C) return;
  int r = i / C, c = i % C;
  float m = -1e30f;
  for (int j = 0; j < ns; ++j)
    m = fmaxf(m, (float)h[((size_t)r * ns + j) * C + c]);
  out[(size_t)r * C + c] = (_Float16)m;
}

// 3 nearest neighbors of each xyz1 point among xyz2, plus normalized 1/d weights
__global__ void knn3_kernel(const float* __restrict__ xyz1, const float* __restrict__ xyz2,
                            int* __restrict__ idx3, float* __restrict__ w3,
                            int N1, int N2, int Bn)
{
  int q = blockIdx.x * blockDim.x + threadIdx.x;
  if (q >= Bn * N1) return;
  int b = q / N1;
  float px = xyz1[q * 3 + 0], py = xyz1[q * 3 + 1], pz = xyz1[q * 3 + 2];
  const float* X2 = xyz2 + (size_t)b * N2 * 3;
  float d0 = 1e30f, d1 = 1e30f, d2 = 1e30f;
  int   i0 = 0, i1 = 0, i2 = 0;
  for (int i = 0; i < N2; ++i) {
    float dx = X2[i * 3 + 0] - px;
    float dy = X2[i * 3 + 1] - py;
    float dz = X2[i * 3 + 2] - pz;
    float d = dx * dx + dy * dy + dz * dz;
    if (d < d0)      { d2 = d1; i2 = i1; d1 = d0; i1 = i0; d0 = d; i0 = i; }
    else if (d < d1) { d2 = d1; i2 = i1; d1 = d; i1 = i; }
    else if (d < d2) { d2 = d; i2 = i; }
  }
  float w0 = 1.0f / (d0 + 1e-8f);
  float w1 = 1.0f / (d1 + 1e-8f);
  float w2 = 1.0f / (d2 + 1e-8f);
  float sw = w0 + w1 + w2;
  idx3[q * 3 + 0] = i0; idx3[q * 3 + 1] = i1; idx3[q * 3 + 2] = i2;
  w3[q * 3 + 0] = w0 / sw; w3[q * 3 + 1] = w1 / sw; w3[q * 3 + 2] = w2 / sw;
}

// row = [points1 | sum_j w_j * points2[idx_j]]  (f16, stride K32, zero-padded)
__global__ void interp_concat_kernel(const _Float16* __restrict__ p1, const _Float16* __restrict__ p2,
                                     const int* __restrict__ idx3, const float* __restrict__ w3,
                                     _Float16* __restrict__ A, int N1, int N2, int C1, int C2,
                                     int K32, int Bn)
{
  int q = blockIdx.x * blockDim.x + threadIdx.x;
  if (q >= Bn * N1) return;
  int b = q / N1;
  _Float16* row = A + (size_t)q * K32;
  if (p1) {
    const _Float16* src = p1 + (size_t)q * C1;
    for (int c = 0; c < C1; ++c) row[c] = src[c];
  }
  const _Float16* P2 = p2 + (size_t)b * N2 * C2;
  int   i0 = idx3[q * 3 + 0], i1 = idx3[q * 3 + 1], i2 = idx3[q * 3 + 2];
  float w0 = w3[q * 3 + 0],  w1 = w3[q * 3 + 1],  w2 = w3[q * 3 + 2];
  for (int c = 0; c < C2; ++c) {
    float v = w0 * (float)P2[(size_t)i0 * C2 + c]
            + w1 * (float)P2[(size_t)i1 * C2 + c]
            + w2 * (float)P2[(size_t)i2 * C2 + c];
    row[C1 + c] = (_Float16)v;
  }
  for (int c = C1 + C2; c < K32; ++c) row[c] = (_Float16)0.0f;
}

// ======================= weight folding into WMMA B-fragment layout =======================
// Packed layout: for (tileN, kblock): 32 lanes x 16 halfs contiguous.
//   lane l: n = tileN*16 + (l&15); halfs j=0..15 hold K = kblock*32 + (l>>4)*16 + j.

__global__ void fold_weights_packed_kernel(const float* __restrict__ W,
                                           const float* __restrict__ gamma,
                                           const float* __restrict__ var,
                                           _Float16* __restrict__ Bp,
                                           int Cin, int K32, int Cout, int nKB)
{
  int i = blockIdx.x * blockDim.x + threadIdx.x;
  if (i >= K32 * Cout) return;
  int j    = i & 15;
  int t    = i >> 4;
  int lane = t & 31;
  int t2   = t >> 5;
  int kb   = t2 % nKB;
  int tN   = t2 / nKB;
  int k = kb * 32 + (lane >> 4) * 16 + j;
  int n = tN * 16 + (lane & 15);
  float s = gamma[n] * rsqrtf(var[n] + EPS_BN);
  Bp[i] = (_Float16)((k < Cin) ? W[(size_t)k * Cout + n] * s : 0.0f);
}

__global__ void fold_bias_kernel(const float* __restrict__ b, const float* __restrict__ beta,
                                 const float* __restrict__ gamma, const float* __restrict__ mean,
                                 const float* __restrict__ var, float* __restrict__ bias2, int Cout)
{
  int c = blockIdx.x * blockDim.x + threadIdx.x;
  if (c >= Cout) return;
  float s = gamma[c] * rsqrtf(var[c] + EPS_BN);
  bias2[c] = (b[c] - mean[c]) * s + beta[c];
}

// ======================= WMMA GEMM: relu( A(MxK32,f16) * Bp + bias ) -> f16 (MxCout) ======
// One wave per 32x32 output tile: 2 M-frags x 2 N-frags, 4 accumulators.
// Per K-step: 8x b128 loads feed 4x v_wmma (2 loads per WMMA).

__global__ void gemm_bn_relu_wmma(const _Float16* __restrict__ A,
                                  const _Float16* __restrict__ Bp,
                                  const float* __restrict__ bias,
                                  _Float16* __restrict__ C,
                                  int M, int K32, int Cout, int tilesN2)
{
  int wave = (blockIdx.x * blockDim.x + threadIdx.x) >> 5;   // wave-uniform
  int lane = threadIdx.x & 31;
  int tilesM2 = (M + 31) >> 5;
  if (wave >= tilesM2 * tilesN2) return;                      // wave-uniform: EXEC all-ones
  int tM2 = wave / tilesN2;
  int tN2 = wave % tilesN2;
  int half = lane >> 4;
  int lrow = lane & 15;
  int nKB = K32 >> 5;

  int row0 = tM2 * 32 + lrow;       if (row0 >= M) row0 = M - 1;   // clamp; store guarded
  int row1 = tM2 * 32 + 16 + lrow;  if (row1 >= M) row1 = M - 1;

  const _Float16* Ap0 = A  + (size_t)row0 * K32 + half * 16;
  const _Float16* Ap1 = A  + (size_t)row1 * K32 + half * 16;
  const _Float16* Bp0 = Bp + ((size_t)(tN2 * 2)     * nKB * 32 + lane) * 16;
  const _Float16* Bp1 = Bp + ((size_t)(tN2 * 2 + 1) * nKB * 32 + lane) * 16;

  v8f acc00 = {}, acc01 = {}, acc10 = {}, acc11 = {};
  for (int kb = 0; kb < nKB; ++kb) {
    v8h a0lo = *(const v8h*)(Ap0);
    v8h a0hi = *(const v8h*)(Ap0 + 8);
    v8h a1lo = *(const v8h*)(Ap1);
    v8h a1hi = *(const v8h*)(Ap1 + 8);
    v8h b0lo = *(const v8h*)(Bp0);
    v8h b0hi = *(const v8h*)(Bp0 + 8);
    v8h b1lo = *(const v8h*)(Bp1);
    v8h b1hi = *(const v8h*)(Bp1 + 8);
    Ap0 += 32; Ap1 += 32; Bp0 += 512; Bp1 += 512;
    v16h av0 = SHUF16(a0lo, a0hi);
    v16h av1 = SHUF16(a1lo, a1hi);
    v16h bv0 = SHUF16(b0lo, b0hi);
    v16h bv1 = SHUF16(b1lo, b1hi);
    acc00 = __builtin_amdgcn_wmma_f32_16x16x32_f16(false, av0, false, bv0, (short)0, acc00, false, false);
    acc01 = __builtin_amdgcn_wmma_f32_16x16x32_f16(false, av0, false, bv1, (short)0, acc01, false, false);
    acc10 = __builtin_amdgcn_wmma_f32_16x16x32_f16(false, av1, false, bv0, (short)0, acc10, false, false);
    acc11 = __builtin_amdgcn_wmma_f32_16x16x32_f16(false, av1, false, bv1, (short)0, acc11, false, false);
  }

  int n0 = tN2 * 32 + lrow;
  int n1 = n0 + 16;
  float bn0 = bias[n0];
  float bn1 = bias[n1];
#pragma unroll
  for (int v = 0; v < 8; ++v) {
    int m0 = tM2 * 32 + v + half * 8;        // rows of upper 16-row band
    int m1 = m0 + 16;                         // rows of lower 16-row band
    if (m0 < M) {
      float x00 = acc00[v] + bn0;
      float x01 = acc01[v] + bn1;
      C[(size_t)m0 * Cout + n0] = (_Float16)(x00 > 0.0f ? x00 : 0.0f);
      C[(size_t)m0 * Cout + n1] = (_Float16)(x01 > 0.0f ? x01 : 0.0f);
    }
    if (m1 < M) {
      float x10 = acc10[v] + bn0;
      float x11 = acc11[v] + bn1;
      C[(size_t)m1 * Cout + n0] = (_Float16)(x10 > 0.0f ? x10 : 0.0f);
      C[(size_t)m1 * Cout + n1] = (_Float16)(x11 > 0.0f ? x11 : 0.0f);
    }
  }
}

// final classifier: out = h @ W2 + b2  (f16 in, f32 out, Cout=13)
__global__ void final_linear_kernel(const _Float16* __restrict__ h, const float* __restrict__ W2,
                                    const float* __restrict__ b2, float* __restrict__ out,
                                    int Np, int Cc, int D)
{
  int i = blockIdx.x * blockDim.x + threadIdx.x;
  if (i >= Np * D) return;
  int n = i / D, d = i % D;
  float s = b2[d];
  const _Float16* hr = h + (size_t)n * Cc;
  for (int c = 0; c < Cc; ++c) s += (float)hr[c] * W2[(size_t)c * D + d];
  out[i] = s;
}

// ======================= host orchestration =======================

extern "C" void kernel_launch(void* const* d_in, const int* in_sizes, int n_in,
                              void* d_out, int out_size, void* d_ws, size_t ws_size,
                              hipStream_t stream)
{
  (void)in_sizes; (void)n_in; (void)out_size; (void)ws_size;
  const int B = 8, N = 8192;

  auto P = [&](int i) -> const float* { return (const float*)d_in[i]; };
  const float* xyz_in = P(0);
  const float* pts_in = P(1);
  const float* W2 = P(2);
  const float* b2 = P(3);

  // param leaves (jax pytree = sorted dict keys): W2,b2, fp1,fp2,fp3,fp4,head1,sa1..sa4;
  // per layer sorted: W, b, beta, gamma, mean, var (6 leaves)
  struct LDef { int base, cin, cout; };
  static const LDef layers[22] = {
    {4,128,128},{10,128,128},{16,128,128},      // fp1   0..2
    {22,320,256},{28,256,128},                  // fp2   3..4
    {34,384,256},{40,256,256},                  // fp3   5..6
    {46,768,256},{52,256,256},                  // fp4   7..8
    {58,128,128},                               // head1 9
    {64,6,32},{70,32,32},{76,32,64},            // sa1   10..12
    {82,67,64},{88,64,64},{94,64,128},          // sa2   13..15
    {100,131,128},{106,128,128},{112,128,256},  // sa3   16..18
    {118,259,256},{124,256,256},{130,256,512},  // sa4   19..21
  };

  // ---- bump allocator over d_ws ----
  char* ws = (char*)d_ws;
  size_t off = 0;
  auto alloc = [&](size_t bytes) -> void* {
    off = (off + 255) & ~(size_t)255;
    void* p = ws + off;
    off += bytes;
    return p;
  };

  float*     XT     = (float*)alloc((size_t)B * N * 3 * 4);
  _Float16*  PT     = (_Float16*)alloc((size_t)B * N * 3 * 2);
  float*     l1x    = (float*)alloc((size_t)B * 1024 * 3 * 4);
  float*     l2x    = (float*)alloc((size_t)B * 256 * 3 * 4);
  float*     l3x    = (float*)alloc((size_t)B * 64 * 3 * 4);
  float*     l4x    = (float*)alloc((size_t)B * 16 * 3 * 4);
  _Float16*  l1p_sa = (_Float16*)alloc((size_t)B * 1024 * 64 * 2);
  _Float16*  l2p_sa = (_Float16*)alloc((size_t)B * 256 * 128 * 2);
  _Float16*  l3p_sa = (_Float16*)alloc((size_t)B * 64 * 256 * 2);
  _Float16*  l4p    = (_Float16*)alloc((size_t)B * 16 * 512 * 2);
  _Float16*  l3p_fp = (_Float16*)alloc((size_t)B * 64 * 256 * 2);
  _Float16*  l2p_fp = (_Float16*)alloc((size_t)B * 256 * 256 * 2);
  _Float16*  l1p_fp = (_Float16*)alloc((size_t)B * 1024 * 128 * 2);
  int*       fidx   = (int*)alloc((size_t)B * 1024 * 4);
  int*       gidx   = (int*)alloc((size_t)B * 1024 * 32 * 4);
  int*       idx3   = (int*)alloc((size_t)B * 8192 * 3 * 4);
  float*     w3     = (float*)alloc((size_t)B * 8192 * 3 * 4);

  void*  WhP[22];
  float* biasP[22];
  int    K32s[22];
  for (int i = 0; i < 22; ++i) {
    int cin = layers[i].cin, cout = layers[i].cout;
    int K32 = ((cin + 31) / 32) * 32;
    K32s[i] = K32;
    WhP[i]   = alloc((size_t)K32 * cout * 2);
    biasP[i] = (float*)alloc((size_t)cout * 4);
  }

  const size_t BIG_ELEMS = (size_t)262144 * 64;   // largest intermediate (SA1 grouped h), f16
  _Float16* BIG0 = (_Float16*)alloc(BIG_ELEMS * 2);
  _Float16* BIG1 = (_Float16*)alloc(BIG_ELEMS * 2);

  auto blocks = [](long n) { return dim3((unsigned)((n + 255) / 256)); };

  // ---- fold all layer weights into packed WMMA B-fragments + f32 bias ----
  for (int i = 0; i < 22; ++i) {
    const LDef& l = layers[i];
    const float* W    = P(l.base + 0);
    const float* b    = P(l.base + 1);
    const float* beta = P(l.base + 2);
    const float* gam  = P(l.base + 3);
    const float* mean = P(l.base + 4);
    const float* var  = P(l.base + 5);
    long nW = (long)K32s[i] * l.cout;
    fold_weights_packed_kernel<<<blocks(nW), 256, 0, stream>>>(
        W, gam, var, (_Float16*)WhP[i], l.cin, K32s[i], l.cout, K32s[i] >> 5);
    fold_bias_kernel<<<blocks(l.cout), 256, 0, stream>>>(b, beta, gam, mean, var, biasP[i], l.cout);
  }

  auto gemm = [&](const _Float16* A, _Float16* Cp, int M, int li) {
    const LDef& l = layers[li];
    int tilesN2 = l.cout / 32;
    long waves = (long)((M + 31) / 32) * tilesN2;
    gemm_bn_relu_wmma<<<dim3((unsigned)((waves + 7) / 8)), 256, 0, stream>>>(
        A, (const _Float16*)WhP[li], biasP[li], Cp, M, K32s[li], l.cout, tilesN2);
  };

  // chain of GEMMs, A0 must live in BIG0; ping-pong BIG1/BIG0; optional final destination
  auto run_chain = [&](const _Float16* A0, int M, int li0, int nlay, _Float16* finalOut) -> _Float16* {
    const _Float16* cur = A0;
    _Float16* outp = BIG1;
    for (int j = 0; j < nlay; ++j) {
      outp = (j + 1 == nlay && finalOut) ? finalOut : ((j & 1) ? BIG0 : BIG1);
      gemm(cur, outp, M, li0 + j);
      cur = outp;
    }
    return outp;
  };

  auto sa_stage = [&](const float* in_xyz, int Nin, const _Float16* in_pts, int Cpts,
                      int S, float radius, int ns, int li0,
                      float* out_xyz, _Float16* out_pts) {
    fps_kernel<<<dim3(B), 256, 0, stream>>>(in_xyz, fidx, Nin, S);
    gather3_kernel<<<blocks((long)B * S * 3), 256, 0, stream>>>(in_xyz, fidx, out_xyz, Nin, S, B);
    ball_query_kernel<<<blocks((long)B * S), 256, 0, stream>>>(in_xyz, out_xyz, gidx, Nin, S,
                                                               radius * radius, ns, B);
    int M = B * S * ns;
    group_kernel<<<blocks((long)M), 256, 0, stream>>>(in_xyz, in_pts, out_xyz, gidx, BIG0,
                                                      Nin, S, ns, Cpts, K32s[li0], B);
    _Float16* last = run_chain(BIG0, M, li0, 3, nullptr);
    int C = layers[li0 + 2].cout;
    maxpool_kernel<<<blocks((long)B * S * C), 256, 0, stream>>>(last, out_pts, B * S, ns, C);
  };

  auto fp_stage = [&](const float* x1, int N1, const float* x2, int N2,
                      const _Float16* p1, int C1, const _Float16* p2, int C2,
                      int li0, int nlay, _Float16* out) -> _Float16* {
    knn3_kernel<<<blocks((long)B * N1), 256, 0, stream>>>(x1, x2, idx3, w3, N1, N2, B);
    interp_concat_kernel<<<blocks((long)B * N1), 256, 0, stream>>>(p1, p2, idx3, w3, BIG0,
                                                                   N1, N2, C1, C2, K32s[li0], B);
    return run_chain(BIG0, B * N1, li0, nlay, out);
  };

  // ---- pipeline ----
  transpose_b3n_f32_kernel<<<blocks((long)B * N * 3), 256, 0, stream>>>(xyz_in, XT, B, N);
  transpose_b3n_f16_kernel<<<blocks((long)B * N * 3), 256, 0, stream>>>(pts_in, PT, B, N);

  sa_stage(XT,  8192, PT,     3,  1024, 0.1f, 32, 10, l1x, l1p_sa);
  sa_stage(l1x, 1024, l1p_sa, 64,  256, 0.2f, 32, 13, l2x, l2p_sa);
  sa_stage(l2x,  256, l2p_sa, 128,  64, 0.4f, 32, 16, l3x, l3p_sa);
  sa_stage(l3x,   64, l3p_sa, 256,  16, 0.8f, 32, 19, l4x, l4p);

  fp_stage(l3x,   64, l4x,   16, l3p_sa, 256, l4p,    512, 7, 2, l3p_fp);
  fp_stage(l2x,  256, l3x,   64, l2p_sa, 128, l3p_fp, 256, 5, 2, l2p_fp);
  fp_stage(l1x, 1024, l2x,  256, l1p_sa,  64, l2p_fp, 256, 3, 2, l1p_fp);
  _Float16* l0p = fp_stage(XT, 8192, l1x, 1024, nullptr, 0, l1p_fp, 128, 0, 3, nullptr);

  // head1 (128->128, WMMA) then final 128->13 classifier
  _Float16* headOut = (l0p == BIG1) ? BIG0 : BIG1;
  gemm(l0p, headOut, B * N, 9);
  final_linear_kernel<<<blocks((long)B * N * 13), 256, 0, stream>>>(
      headOut, W2, b2, (float*)d_out, B * N, 128, 13);
}
